// BidirectionalMamba_78039555769090
// MI455X (gfx1250) — matrix operand order repaired and checked by reading, not compiled, not run on hardware
//
#include <hip/hip_runtime.h>
#include <hip/hip_bf16.h>
#include <math.h>

// ---------------------------------------------------------------------------
// Types for CDNA5 WMMA
// ---------------------------------------------------------------------------
typedef __attribute__((ext_vector_type(16))) __bf16 v16bf;
typedef __attribute__((ext_vector_type(8)))  float  v8f;

union FragBF { v16bf v; unsigned u[8]; };

static __device__ __forceinline__ unsigned pack_bf16(float a, float b) {
    union { __bf16 h[2]; unsigned u; } r;
    r.h[0] = (__bf16)a; r.h[1] = (__bf16)b;
    return r.u;
}
static __device__ __forceinline__ unsigned short bf16_bits(float x) {
    union { __bf16 h; unsigned short u; } r; r.h = (__bf16)x; return r.u;
}
static __device__ __forceinline__ float softplus_f(float x) {
    return (x > 20.0f) ? x : log1pf(__expf(x));
}
static __device__ __forceinline__ float silu_f(float x) {
    return x / (1.0f + __expf(-x));
}

// Async global->LDS copy of 16B per lane (CDNA5, tracked by ASYNCcnt).
// The LDS pointer is passed BOTH as the address operand (low 32 bits of the
// flat address == LDS byte offset, ISA sec.10.2 LDS aperture) AND as an escape
// operand so alias analysis knows the asm writes the __shared__ object.
static __device__ __forceinline__ void async_b128(void* lds_ptr, const void* gaddr) {
    unsigned lds_off = (unsigned)(size_t)lds_ptr;
    asm volatile("global_load_async_to_lds_b128 %0, %1, off"
                 :: "v"(lds_off), "v"(gaddr), "v"(lds_ptr) : "memory");
}
static __device__ __forceinline__ void wait_async_le4() {
    asm volatile("s_wait_asynccnt 0x4" ::: "memory");
}
static __device__ __forceinline__ void wait_async_0() {
    asm volatile("s_wait_asynccnt 0x0" ::: "memory");
}

// ---------------------------------------------------------------------------
// fp32 -> bf16 bulk convert (4 elements / thread; n must be multiple of 1024)
// ---------------------------------------------------------------------------
__global__ __launch_bounds__(256) void cvt_bf16_kernel(
    const float* __restrict__ s, unsigned* __restrict__ d)
{
    size_t i = (size_t)blockIdx.x * 256 + threadIdx.x;
    float4 v = *(const float4*)(s + i * 4);
    d[i * 2]     = pack_bf16(v.x, v.y);
    d[i * 2 + 1] = pack_bf16(v.z, v.w);
}

// xdbl[:, :64] fp32 (stride 128) -> bf16 [8192,64]
__global__ __launch_bounds__(256) void cvt_xdbl_kernel(
    const float* __restrict__ xdbl, unsigned* __restrict__ out)
{
    int i = blockIdx.x * 256 + threadIdx.x;       // pair id, total 8192*32
    int row = i >> 5, p = i & 31;
    float2 v = *(const float2*)(xdbl + (size_t)row * 128 + p * 2);
    out[(size_t)row * 32 + p] = pack_bf16(v.x, v.y);
}

// ---------------------------------------------------------------------------
// bf16 WMMA GEMM:  C[m,n] = sum_k A[m,k] * W[n,k]   (+ epilogue), fp32 out.
// A: bf16 [M,K] (lda), optionally time-reversed rows (SEQ=1024 blocks)
// W: bf16 [N,K] (ldw);  C: fp32 [M,N] (ldc)
// 128x128x32 tiles, 256 threads = 8 wave32 waves, 4x2 WMMA tiles per wave.
// Staging: double-buffered GLOBAL_LOAD_ASYNC_TO_LDS_B128 (4 ops/wave/tile).
// epilogue: 0 = none, 1 = softplus(acc + bias[n]), 2 = acc + bias[n]
// ---------------------------------------------------------------------------
#define BM 128
#define BN 128
#define BK 32

__global__ __launch_bounds__(256) void gemm_wmma_bf16(
    const unsigned short* __restrict__ A, int lda,
    const unsigned short* __restrict__ W, int ldw,
    float* __restrict__ C, int ldc,
    int K, int revA,
    int epilogue, const float* __restrict__ bias)
{
    // [buf][A(10240B) | W(10240B)] ; rows pitched 80B (64B data + 16B pad)
    __shared__ unsigned lds_u32[2 * 2 * 2560];

    const int tid  = threadIdx.x;
    const int lane = tid & 31;
    const int wave = tid >> 5;
    const int wm   = wave & 1;
    const int wn   = wave >> 1;
    const int blockN = blockIdx.x * BN;
    const int blockM = blockIdx.y * BM;

    auto issue = [&](int k0, int buf) {
        char* base = (char*)lds_u32 + buf * 20480;
        #pragma unroll
        for (int j = 0; j < 2; ++j) {
            int q = tid + j * 256;                 // 0..511
            int row = q >> 2, s = q & 3;           // 4 x 16B segs per row
            int gr = blockM + row;
            if (revA) gr = (gr & ~1023) + (1023 - (gr & 1023));
            async_b128(base + row * 80 + s * 16,
                       A + (size_t)gr * lda + k0 + s * 8);
        }
        #pragma unroll
        for (int j = 0; j < 2; ++j) {
            int q = tid + j * 256;
            int row = q >> 2, s = q & 3;
            async_b128(base + 10240 + row * 80 + s * 16,
                       W + (size_t)(blockN + row) * ldw + k0 + s * 8);
        }
    };

    v8f acc[4][2] = {};

    issue(0, 0);
    int buf = 0;
    for (int k0 = 0; k0 < K; k0 += BK, buf ^= 1) {
        if (k0 + BK < K) { issue(k0 + BK, buf ^ 1); wait_async_le4(); }
        else             { wait_async_0(); }
        __syncthreads();

        const unsigned* Ab = lds_u32 + buf * 5120;
        const unsigned* Wb = Ab + 2560;
        const int lg = lane >> 4;                 // lane group 0/1

        FragBF af[4], bfm[2];
        #pragma unroll
        for (int tm = 0; tm < 4; ++tm) {
            int row = wm * 64 + tm * 16 + (lane & 15);       // M index
            #pragma unroll
            for (int v = 0; v < 8; ++v) {
                // A 16x32 bf16 layout: K-pair index = (v>>2)*8 + lg*4 + (v&3)
                int p = ((v >> 2) << 3) + (lg << 2) + (v & 3);
                af[tm].u[v] = Ab[row * 20 + p];
            }
        }
        #pragma unroll
        for (int tn = 0; tn < 2; ++tn) {
            int nrow = wn * 32 + tn * 16 + (lane & 15);      // N index
            #pragma unroll
            for (int j = 0; j < 8; ++j)
                // B 32x16 bf16 layout: lane = N, K-pair index = (j>>2)*8 + lg*4 + (j&3)
                bfm[tn].u[j] = Wb[nrow * 20 + ((j >> 2) << 3) + (lg << 2) + (j & 3)];
        }
        #pragma unroll
        for (int tm = 0; tm < 4; ++tm)
            #pragma unroll
            for (int tn = 0; tn < 2; ++tn)
                acc[tm][tn] = __builtin_amdgcn_wmma_f32_16x16x32_bf16(
                    false, af[tm].v, false, bfm[tn].v,
                    (short)0, acc[tm][tn], false, false);
        __syncthreads();
    }

    // D layout: VGPR r, lane L -> M = r + (L>>4)*8, N = L&15
    #pragma unroll
    for (int tm = 0; tm < 4; ++tm) {
        #pragma unroll
        for (int tn = 0; tn < 2; ++tn) {
            int mbase = blockM + wm * 64 + tm * 16 + ((lane >> 4) << 3);
            int n     = blockN + wn * 32 + tn * 16 + (lane & 15);
            #pragma unroll
            for (int r = 0; r < 8; ++r) {
                float v = acc[tm][tn][r];
                if (epilogue == 1)      v = softplus_f(v + bias[n]);
                else if (epilogue == 2) v = v + bias[n];
                C[(size_t)(mbase + r) * ldc + n] = v;
            }
        }
    }
}

// ---------------------------------------------------------------------------
// Causal depthwise conv (width 4) + bias + SiLU.
// u lives in xz[:, 0:2048] (row stride 4096). Emits fp32 (scan) + bf16 (GEMM).
// ---------------------------------------------------------------------------
__global__ __launch_bounds__(256) void conv_silu_kernel(
    const float* __restrict__ xz,
    const float* __restrict__ convw,   // [2048,4]
    const float* __restrict__ convb,   // [2048]
    float* __restrict__ ucf,
    unsigned short* __restrict__ ucbf)
{
    int idx = blockIdx.x * 256 + threadIdx.x;     // 0 .. 8192*2048-1
    int c   = idx & 2047;
    int row = idx >> 11;                          // b*1024 + l
    int l   = row & 1023;
    float4 w = *(const float4*)(convw + c * 4);
    float acc = convb[c];
    if (l >= 3) acc += xz[(size_t)(row - 3) * 4096 + c] * w.x;
    if (l >= 2) acc += xz[(size_t)(row - 2) * 4096 + c] * w.y;
    if (l >= 1) acc += xz[(size_t)(row - 1) * 4096 + c] * w.z;
    acc += xz[(size_t)row * 4096 + c] * w.w;
    float v = silu_f(acc);
    ucf[idx]  = v;
    ucbf[idx] = bf16_bits(v);
}

// ---------------------------------------------------------------------------
// Selective scan. Lane = channel, 32 states in VGPRs.
// B/C chunks (64 steps) double-buffered via async global->LDS.
// Fused epilogue: y = (scan_y + uc*Dp) * silu(z), stored as bf16 for Wout GEMM.
// ---------------------------------------------------------------------------
__global__ __launch_bounds__(256) void scan_kernel(
    const float* __restrict__ dt,    // [8192,2048]
    const float* __restrict__ ucf,   // [8192,2048]
    const float* __restrict__ xdbl,  // [8192,128]: B at +64, C at +96
    const float* __restrict__ xz,    // z at col 2048, row stride 4096
    const float* __restrict__ Alog,  // [2048,32]
    const float* __restrict__ Dp,    // [2048]
    unsigned short* __restrict__ ybf)// [8192,2048] bf16
{
    // [buf][ B 8192B | C 8192B ]
    __shared__ unsigned char slds[2 * 16384];
    const int tid = threadIdx.x;
    const int b   = blockIdx.x >> 3;
    const int d   = ((blockIdx.x & 7) << 8) + tid;

    auto issue = [&](int chunk, int buf) {
        unsigned char* base = slds + buf * 16384;
        #pragma unroll
        for (int j = 0; j < 2; ++j) {
            int q = tid + j * 256;                 // 0..511
            int t = q >> 3, s = q & 7;             // 8 x 16B segs per 32-fp row
            size_t row = (size_t)b * 1024 + chunk * 64 + t;
            async_b128(base + t * 128 + s * 16,          xdbl + row * 128 + 64 + s * 4);
            async_b128(base + 8192 + t * 128 + s * 16,   xdbl + row * 128 + 96 + s * 4);
        }
    };

    float a[32], h[32];
    #pragma unroll
    for (int n = 0; n < 32; ++n) {
        a[n] = -__expf(Alog[d * 32 + n]);
        h[n] = 0.0f;
    }
    const float dp = Dp[d];

    issue(0, 0);
    int buf = 0;
    for (int chunk = 0; chunk < 16; ++chunk, buf ^= 1) {
        if (chunk < 15) { issue(chunk + 1, buf ^ 1); wait_async_le4(); }
        else            { wait_async_0(); }
        __syncthreads();
        const float* Bs = (const float*)(slds + buf * 16384);
        const float* Cs = Bs + 2048;
        for (int t = 0; t < 64; ++t) {
            const size_t row = (size_t)b * 1024 + chunk * 64 + t;
            const float dtv = dt [row * 2048 + d];
            const float ucv = ucf[row * 2048 + d];
            const float zv  = xz [row * 4096 + 2048 + d];
            const float du  = dtv * ucv;
            float yv = 0.0f;
            #pragma unroll
            for (int n = 0; n < 32; ++n) {
                float dA = __expf(dtv * a[n]);
                h[n] = dA * h[n] + du * Bs[t * 32 + n];
                yv  += h[n] * Cs[t * 32 + n];
            }
            yv = (yv + ucv * dp) * silu_f(zv);
            ybf[row * 2048 + d] = bf16_bits(yv);
        }
        __syncthreads();
    }
}

// ---------------------------------------------------------------------------
// Per-token LayerNorm over 1024 features -> bf16 into one half of comb
// (row stride 2048), optional time-reversal of the output row.
// ---------------------------------------------------------------------------
__global__ __launch_bounds__(256) void ln_kernel(
    const float* __restrict__ in_,   // [8192,1024]
    const float* __restrict__ g,
    const float* __restrict__ bta,
    unsigned short* __restrict__ out, // combbf + dir*1024, stride 2048
    int rev)
{
    __shared__ float sbuf[256], ssbuf[256];
    const int row = blockIdx.x;
    const int tid = threadIdx.x;
    const float* x = in_ + (size_t)row * 1024;

    float s = 0.0f, ss = 0.0f;
    #pragma unroll
    for (int i = 0; i < 4; ++i) {
        float v = x[tid + i * 256];
        s += v; ss += v * v;
    }
    sbuf[tid] = s; ssbuf[tid] = ss;
    __syncthreads();
    for (int off = 128; off > 0; off >>= 1) {
        if (tid < off) { sbuf[tid] += sbuf[tid + off]; ssbuf[tid] += ssbuf[tid + off]; }
        __syncthreads();
    }
    const float mean = sbuf[0] * (1.0f / 1024.0f);
    const float var  = ssbuf[0] * (1.0f / 1024.0f) - mean * mean;
    const float rstd = rsqrtf(var + 1e-5f);

    int orow = row;
    if (rev) orow = (row & ~1023) + (1023 - (row & 1023));
    unsigned short* o = out + (size_t)orow * 2048;
    #pragma unroll
    for (int i = 0; i < 4; ++i) {
        int j = tid + i * 256;
        o[j] = bf16_bits((x[j] - mean) * rstd * g[j] + bta[j]);
    }
}

// ---------------------------------------------------------------------------
// Host launcher
// ---------------------------------------------------------------------------
extern "C" void kernel_launch(void* const* d_in, const int* in_sizes, int n_in,
                              void* d_out, int out_size, void* d_ws, size_t ws_size,
                              hipStream_t stream) {
    (void)in_sizes; (void)n_in; (void)out_size; (void)ws_size;

    const int M = 8192;            // BATCH * SEQ
    const float* x = (const float*)d_in[0];

    // Workspace layout (256B-aligned slices)
    size_t off = 0;
    auto alloc = [&](size_t bytes) {
        void* p = (char*)d_ws + off;
        off += (bytes + 255) & ~(size_t)255;
        return p;
    };
    float*          xz      = (float*)alloc((size_t)M * 4096 * 4); // u|z ; reused as raw-out tmp
    float*          ucf     = (float*)alloc((size_t)M * 2048 * 4);
    float*          dtb     = (float*)alloc((size_t)M * 2048 * 4);
    float*          xdbl    = (float*)alloc((size_t)M * 128 * 4);
    unsigned short* xbf     = (unsigned short*)alloc((size_t)M * 1024 * 2);
    unsigned short* ucbf    = (unsigned short*)alloc((size_t)M * 2048 * 2);
    unsigned short* xdblbf  = (unsigned short*)alloc((size_t)M * 64 * 2);
    unsigned short* ybf     = (unsigned short*)alloc((size_t)M * 2048 * 2);
    unsigned short* combbf  = (unsigned short*)alloc((size_t)M * 2048 * 2);
    unsigned short* Winbf   = (unsigned short*)alloc((size_t)4096 * 1024 * 2);
    unsigned short* Wxbf    = (unsigned short*)alloc((size_t)128 * 2048 * 2);
    unsigned short* Wdtbf   = (unsigned short*)alloc((size_t)2048 * 64 * 2);
    unsigned short* Woutbf  = (unsigned short*)alloc((size_t)1024 * 2048 * 2);
    unsigned short* Wfusebf = (unsigned short*)alloc((size_t)1024 * 2048 * 2);

    // One-shot conversions shared by both directions
    cvt_bf16_kernel<<<(M * 1024) / 1024, 256, 0, stream>>>(x, (unsigned*)xbf);
    cvt_bf16_kernel<<<(1024 * 2048) / 1024, 256, 0, stream>>>(
        (const float*)d_in[23], (unsigned*)Wfusebf);

    for (int dir = 0; dir < 2; ++dir) {
        const float* Win   = (const float*)d_in[1  + dir * 11];
        const float* convw = (const float*)d_in[2  + dir * 11];
        const float* convb = (const float*)d_in[3  + dir * 11];
        const float* Wx    = (const float*)d_in[4  + dir * 11];
        const float* Wdt   = (const float*)d_in[5  + dir * 11];
        const float* bdt   = (const float*)d_in[6  + dir * 11];
        const float* Alog  = (const float*)d_in[7  + dir * 11];
        const float* Dp    = (const float*)d_in[8  + dir * 11];
        const float* Wout  = (const float*)d_in[9  + dir * 11];
        const float* ln_g  = (const float*)d_in[10 + dir * 11];
        const float* ln_b  = (const float*)d_in[11 + dir * 11];

        // weights -> bf16
        cvt_bf16_kernel<<<(4096 * 1024) / 1024, 256, 0, stream>>>(Win,  (unsigned*)Winbf);
        cvt_bf16_kernel<<<(128 * 2048)  / 1024, 256, 0, stream>>>(Wx,   (unsigned*)Wxbf);
        cvt_bf16_kernel<<<(2048 * 64)   / 1024, 256, 0, stream>>>(Wdt,  (unsigned*)Wdtbf);
        cvt_bf16_kernel<<<(1024 * 2048) / 1024, 256, 0, stream>>>(Wout, (unsigned*)Woutbf);

        // xz = x @ Win^T  [8192,4096]; backward reads x time-reversed
        gemm_wmma_bf16<<<dim3(4096 / BN, M / BM), 256, 0, stream>>>(
            xbf, 1024, Winbf, 1024, xz, 4096, 1024, dir, 0, nullptr);

        // uc = silu(conv(u) + convb)  (fp32 + bf16)
        conv_silu_kernel<<<(M * 2048) / 256, 256, 0, stream>>>(xz, convw, convb, ucf, ucbf);

        // x_dbl = uc @ Wx^T  [8192,128]
        gemm_wmma_bf16<<<dim3(128 / BN, M / BM), 256, 0, stream>>>(
            ucbf, 2048, Wxbf, 2048, xdbl, 128, 2048, 0, 0, nullptr);

        // bf16 slice of x_dbl[:, :64] for the dt GEMM
        cvt_xdbl_kernel<<<(M * 32) / 256, 256, 0, stream>>>(xdbl, (unsigned*)xdblbf);

        // dt = softplus(x_dbl[:, :64] @ Wdt^T + bdt)  [8192,2048]
        gemm_wmma_bf16<<<dim3(2048 / BN, M / BM), 256, 0, stream>>>(
            xdblbf, 64, Wdtbf, 64, dtb, 2048, 64, 0, 1, bdt);

        // selective scan (+ fused uc*Dp, silu(z)); y -> bf16
        scan_kernel<<<64, 256, 0, stream>>>(dtb, ucf, xdbl, xz, Alog, Dp, ybf);

        // raw = y @ Wout^T  [8192,1024] into xz region (z is dead now)
        gemm_wmma_bf16<<<dim3(1024 / BN, M / BM), 256, 0, stream>>>(
            ybf, 2048, Woutbf, 2048, xz, 1024, 2048, 0, 0, nullptr);

        // LayerNorm -> bf16 comb half; backward dir un-reverses rows
        ln_kernel<<<M, 256, 0, stream>>>(xz, ln_g, ln_b, combbf + dir * 1024, dir);
    }

    // out = comb @ Wfuse^T + bfuse  [8192,1024]
    const float* bfuse = (const float*)d_in[24];
    gemm_wmma_bf16<<<dim3(1024 / BN, M / BM), 256, 0, stream>>>(
        combbf, 2048, Wfusebf, 2048, (float*)d_out, 1024, 2048, 0, 2, bfuse);
}
